// PairformerLayer_89910845374471
// MI455X (gfx1250) — matrix operand order repaired and checked by reading, not compiled, not run on hardware
//
#include <hip/hip_runtime.h>
#include <hip/hip_bf16.h>

// ---------------------------------------------------------------------------
// Pairformer layer for gfx1250 (MI455X).
// - All matmuls via v_wmma_f32_16x16x32_bf16 (f32 accumulate).
// - Weights pre-converted to bf16 once per call -> B fragments are single
//   aligned 32B loads (no v_cvt storm on the WMMA critical path).
// - LDS staging uses CDNA5 async loads (global_load_async_to_lds_b128,
//   ASYNCcnt + s_wait_asynccnt) via inline asm.
// - z (32MB fp32) fits in the 192MB L2 -> multi-kernel pipeline through a
//   global workspace costs L2 bandwidth, not HBM.
// ---------------------------------------------------------------------------

typedef __attribute__((ext_vector_type(16))) __bf16 v16bf;
typedef __attribute__((ext_vector_type(2)))  __bf16 bf16x2;
typedef __attribute__((ext_vector_type(8)))  float  v8f;

#define LN_EPS    1e-5f
#define ATT_SCALE 0.17677669529663687f   // 1/sqrt(32)

static __device__ __forceinline__ v8f wmma_bf16(v16bf a, v16bf b, v8f c) {
  return __builtin_amdgcn_wmma_f32_16x16x32_bf16(false, a, false, b, (short)0, c, false, false);
}

static __device__ __forceinline__ float sigmoidf_(float x) { return 1.0f / (1.0f + __expf(-x)); }

// CDNA5 async memory->LDS copy (16B per lane), tracked by ASYNCcnt.
static __device__ __forceinline__ void async_ld_b128(void* lds_dst, const void* gsrc) {
  const unsigned lds_off = (unsigned)(unsigned long long)(uintptr_t)lds_dst;  // low 32b = LDS offset
  asm volatile("global_load_async_to_lds_b128 %0, %1, off"
               :: "v"(lds_off), "v"((unsigned long long)(uintptr_t)gsrc)
               : "memory");
}
static __device__ __forceinline__ void wait_async0() {
  asm volatile("s_wait_asynccnt 0" ::: "memory");
}

// A fragment (16x32 bf16) from a row-major bf16 tile with given row stride.
// Layout: lane m = lane&15, half = lane>>4; VGPR g holds K = 2*g + 16*half (+0/1).
static __device__ __forceinline__ v16bf load_a(const __bf16* tile, int stride, int kk, int lane) {
  const int half = (lane >> 4) & 1;
  const int m = lane & 15;
  const __bf16* row = tile + m * stride + kk + 16 * half;
  v16bf a;
#pragma unroll
  for (int g = 0; g < 8; ++g) {
    const int k0 = g << 1;
    bf16x2 t = *(const bf16x2*)(row + k0);
    a[2 * g] = t[0];
    a[2 * g + 1] = t[1];
  }
  return a;
}

// B fragment (32x16): lane n = lane&15 -> column colBase+n; element e -> K = kk + 16*half + e.
// Rows contiguous in K -> single aligned 32B vector load.
static __device__ __forceinline__ v16bf load_b_bf16rows(const __bf16* Bm, int ldk, int colBase, int kk, int lane) {
  const int n = lane & 15, half = (lane >> 4) & 1;
  return *(const v16bf*)(Bm + (size_t)(colBase + n) * ldk + kk + 16 * half);
}

static __device__ __forceinline__ v16bf load_b_bf16rows_guard(const __bf16* Bm, int ldk, int colBase,
                                                              int kk, int lane, int nrows) {
  const int n = lane & 15, half = (lane >> 4) & 1;
  if (colBase + n < nrows)
    return *(const v16bf*)(Bm + (size_t)(colBase + n) * ldk + kk + 16 * half);
  v16bf b;
#pragma unroll
  for (int e = 0; e < 16; ++e) b[e] = (__bf16)0.0f;
  return b;
}

// B fragment for P @ V with V stored [krow][32]: B[K=kidx][N=dh].
static __device__ __forceinline__ v16bf load_b_v(const __bf16* vs, int kBase, int dhBase, int lane) {
  const int n = lane & 15, half = (lane >> 4) & 1;
  v16bf b;
#pragma unroll
  for (int e = 0; e < 16; ++e) b[e] = vs[(size_t)(kBase + e + 16 * half) * 32 + dhBase + n];
  return b;
}

// LayerNorm 128 rows (each 128 wide) -> bf16 LDS tile. One wave per row, shfl reduction.
static __device__ __forceinline__ void ln_rows_to_lds(const float* __restrict__ src,
                                                      const float* __restrict__ gw,
                                                      const float* __restrict__ gb,
                                                      __bf16 (*xs)[136], int rowBase, int transpose,
                                                      int lane, int wave) {
  for (int rr = wave; rr < 128; rr += 8) {
    const int r0 = rowBase + rr;
    const int i0 = r0 >> 8, j0 = r0 & 255;
    const float* p = src + (size_t)(transpose ? (j0 * 256 + i0) : r0) * 128;
    float4 xv = ((const float4*)p)[lane];
    float s  = xv.x + xv.y + xv.z + xv.w;
    float sq = xv.x * xv.x + xv.y * xv.y + xv.z * xv.z + xv.w * xv.w;
#pragma unroll
    for (int off = 16; off > 0; off >>= 1) { s += __shfl_xor(s, off); sq += __shfl_xor(sq, off); }
    const float mean = s * (1.0f / 128.0f);
    const float inv = rsqrtf(sq * (1.0f / 128.0f) - mean * mean + LN_EPS);
    const int c = lane * 4;
    xs[rr][c + 0] = (__bf16)((xv.x - mean) * inv * gw[c + 0] + gb[c + 0]);
    xs[rr][c + 1] = (__bf16)((xv.y - mean) * inv * gw[c + 1] + gb[c + 1]);
    xs[rr][c + 2] = (__bf16)((xv.z - mean) * inv * gw[c + 2] + gb[c + 2]);
    xs[rr][c + 3] = (__bf16)((xv.w - mean) * inv * gw[c + 3] + gb[c + 3]);
  }
}

// One 16x16 output tile, K=128, A from LDS, B from bf16 weights.
static __device__ __forceinline__ v8f gemm_k128(const __bf16 (*xs)[136], int mBase,
                                                const __bf16* W, int colBase, int lane) {
  v8f acc = {0.f, 0.f, 0.f, 0.f, 0.f, 0.f, 0.f, 0.f};
#pragma unroll
  for (int kk = 0; kk < 128; kk += 32) {
    v16bf a = load_a(&xs[mBase][0], 136, kk, lane);
    v16bf b = load_b_bf16rows(W, 128, colBase, kk, lane);
    acc = wmma_bf16(a, b, acc);
  }
  return acc;
}

// ---------------------------------------------------------------------------
// weight fp32 -> bf16 pre-conversion (one launch, all 23 tensors)
// ---------------------------------------------------------------------------
struct WPack {
  const float* src[23];
  unsigned len[23];
  unsigned off[23];
};

__global__ __launch_bounds__(256) void k_cvt_w(WPack p, __bf16* __restrict__ dst) {
  const int s = blockIdx.y;
  const float* __restrict__ src = p.src[s];
  __bf16* __restrict__ d = dst + p.off[s];
  const unsigned n = p.len[s];
  for (unsigned t = blockIdx.x * 256 + threadIdx.x; t < n; t += gridDim.x * 256)
    d[t] = (__bf16)src[t];
}

// ---------------------------------------------------------------------------
// tri_mult stage 1: LN + gated projections -> a/b planes (d-major) + output gate
// ---------------------------------------------------------------------------
__global__ __launch_bounds__(256) void k_trimult_in(
    const float* __restrict__ z, const float* __restrict__ mask,
    const float* __restrict__ niw, const float* __restrict__ nib,
    const __bf16* __restrict__ p_in, const __bf16* __restrict__ g_in, const __bf16* __restrict__ g_out,
    __bf16* __restrict__ aT, __bf16* __restrict__ bT, __bf16* __restrict__ gsig, int transpose_store) {
  __shared__ __align__(32) __bf16 xs[128][136];
  const int lane = threadIdx.x & 31, wave = threadIdx.x >> 5;
  const int rowBase = blockIdx.x * 128;
  ln_rows_to_lds(z, niw, nib, xs, rowBase, 0, lane, wave);
  __syncthreads();
  const int mBase = wave * 16;
  const int n = lane & 15, half = (lane >> 4) & 1;
#pragma unroll 1
  for (int c = 0; c < 16; ++c) {
    const int colBase = c * 16;
    v8f ap = gemm_k128(xs, mBase, p_in, colBase, lane);
    v8f ag = gemm_k128(xs, mBase, g_in, colBase, lane);
    const int d = colBase + n;
    __bf16* plane = (d < 128) ? (aT + (size_t)d * 65536) : (bT + (size_t)(d - 128) * 65536);
#pragma unroll
    for (int r = 0; r < 8; ++r) {
      const int r0 = rowBase + mBase + r + 8 * half;
      const float val = ap[r] * sigmoidf_(ag[r]) * mask[r0];
      const int i0 = r0 >> 8, j0 = r0 & 255;
      plane[transpose_store ? (j0 * 256 + i0) : r0] = (__bf16)val;
    }
  }
#pragma unroll 1
  for (int c = 0; c < 8; ++c) {
    const int colBase = c * 16;
    v8f ag = gemm_k128(xs, mBase, g_out, colBase, lane);
    const int d = colBase + n;
#pragma unroll
    for (int r = 0; r < 8; ++r) {
      const int r0 = rowBase + mBase + r + 8 * half;
      gsig[(size_t)r0 * 128 + d] = (__bf16)sigmoidf_(ag[r]);
    }
  }
}

// ---------------------------------------------------------------------------
// tri_mult stage 2: per-channel GEMM  O_d = A_d (256x256) @ B_d^T
// ---------------------------------------------------------------------------
__global__ __launch_bounds__(256) void k_tri_gemm(
    const __bf16* __restrict__ aT, const __bf16* __restrict__ bT, float* __restrict__ o) {
  __shared__ __align__(32) __bf16 as[16][264];
  const int d = blockIdx.x;
  const int rt = blockIdx.y;
  const int lane = threadIdx.x & 31, wave = threadIdx.x >> 5;
  const __bf16* A = aT + (size_t)d * 65536 + (size_t)rt * 16 * 256;
  const __bf16* Bm = bT + (size_t)d * 65536;
  // async stage A (16x256 bf16) into LDS: 512 x 16B transfers
  for (int t = threadIdx.x; t < 512; t += 256) {
    const int rr = t >> 5, cc = (t & 31) * 8;
    async_ld_b128(&as[rr][cc], A + rr * 256 + cc);
  }
  wait_async0();
  __syncthreads();
  const int n = lane & 15, half = (lane >> 4) & 1;
#pragma unroll 1
  for (int cti = 0; cti < 2; ++cti) {
    const int jBase = (wave + cti * 8) * 16;
    v8f acc = {0.f, 0.f, 0.f, 0.f, 0.f, 0.f, 0.f, 0.f};
#pragma unroll
    for (int kk = 0; kk < 256; kk += 32) {
      v16bf a = load_a(&as[0][0], 264, kk, lane);
      v16bf b = load_b_bf16rows(Bm, 256, jBase, kk, lane);
      acc = wmma_bf16(a, b, acc);
    }
#pragma unroll
    for (int r = 0; r < 8; ++r) {
      const int i = rt * 16 + r + 8 * half;
      const int j = jBase + n;
      o[((size_t)i * 256 + j) * 128 + d] = acc[r];
    }
  }
}

// ---------------------------------------------------------------------------
// tri_mult stage 3: z' = zres + LN(o) @ p_out^T * gsig
// ---------------------------------------------------------------------------
__global__ __launch_bounds__(256) void k_trimult_out(
    const float* __restrict__ o, const __bf16* __restrict__ gsig,
    const float* __restrict__ now, const float* __restrict__ nob,
    const __bf16* __restrict__ p_out, const float* __restrict__ zres, float* __restrict__ zout) {
  __shared__ __align__(32) __bf16 xs[128][136];
  const int lane = threadIdx.x & 31, wave = threadIdx.x >> 5;
  const int rowBase = blockIdx.x * 128;
  ln_rows_to_lds(o, now, nob, xs, rowBase, 0, lane, wave);
  __syncthreads();
  const int mBase = wave * 16;
  const int n = lane & 15, half = (lane >> 4) & 1;
#pragma unroll 1
  for (int c = 0; c < 8; ++c) {
    const int colBase = c * 16;
    v8f acc = gemm_k128(xs, mBase, p_out, colBase, lane);
    const int d = colBase + n;
#pragma unroll
    for (int r = 0; r < 8; ++r) {
      const int r0 = rowBase + mBase + r + 8 * half;
      const size_t idx = (size_t)r0 * 128 + d;
      zout[idx] = zres[idx] + acc[r] * (float)gsig[idx];
    }
  }
}

// ---------------------------------------------------------------------------
// attention stage 1: LN + q,k,v,gate,bias projections
// ---------------------------------------------------------------------------
__global__ __launch_bounds__(256) void k_attn_qkvg(
    const float* __restrict__ zc, const float* __restrict__ nw, const float* __restrict__ nb,
    const __bf16* __restrict__ wq, const __bf16* __restrict__ wk, const __bf16* __restrict__ wv,
    const __bf16* __restrict__ wg, const __bf16* __restrict__ wbias,
    __bf16* __restrict__ qw, __bf16* __restrict__ kw, __bf16* __restrict__ vw,
    __bf16* __restrict__ gs, float* __restrict__ tb, int transpose) {
  __shared__ __align__(32) __bf16 xs[128][136];
  const int lane = threadIdx.x & 31, wave = threadIdx.x >> 5;
  const int rowBase = blockIdx.x * 128;
  ln_rows_to_lds(zc, nw, nb, xs, rowBase, transpose, lane, wave);
  __syncthreads();
  const int mBase = wave * 16;
  const int n = lane & 15, half = (lane >> 4) & 1;
#pragma unroll 1
  for (int which = 0; which < 3; ++which) {
    const __bf16* W = (which == 0) ? wq : ((which == 1) ? wk : wv);
    __bf16* dst = (which == 0) ? qw : ((which == 1) ? kw : vw);
    const float scale = (which == 0) ? ATT_SCALE : 1.0f;
#pragma unroll 1
    for (int c = 0; c < 8; ++c) {
      const int colBase = c * 16;
      v8f acc = gemm_k128(xs, mBase, W, colBase, lane);
      const int col = colBase + n, h = col >> 5, dh = col & 31;
#pragma unroll
      for (int r = 0; r < 8; ++r) {
        const int r0 = rowBase + mBase + r + 8 * half;
        const int i0 = r0 >> 8, j0 = r0 & 255;
        dst[((size_t)(i0 * 4 + h) * 256 + j0) * 32 + dh] = (__bf16)(acc[r] * scale);
      }
    }
  }
#pragma unroll 1
  for (int c = 0; c < 8; ++c) {
    const int colBase = c * 16;
    v8f acc = gemm_k128(xs, mBase, wg, colBase, lane);
    const int col = colBase + n;
#pragma unroll
    for (int r = 0; r < 8; ++r) {
      const int r0 = rowBase + mBase + r + 8 * half;
      gs[(size_t)r0 * 128 + col] = (__bf16)sigmoidf_(acc[r]);
    }
  }
  {
    v8f acc = {0.f, 0.f, 0.f, 0.f, 0.f, 0.f, 0.f, 0.f};
#pragma unroll
    for (int kk = 0; kk < 128; kk += 32) {
      v16bf a = load_a(&xs[mBase][0], 136, kk, lane);
      v16bf b = load_b_bf16rows_guard(wbias, 128, 0, kk, lane, 4);
      acc = wmma_bf16(a, b, acc);
    }
    if (n < 4) {
#pragma unroll
      for (int r = 0; r < 8; ++r) {
        const int r0 = rowBase + mBase + r + 8 * half;
        const int i0 = r0 >> 8, j0 = r0 & 255;
        tb[((size_t)n * 256 + i0) * 256 + j0] = acc[r];
      }
    }
  }
}

// ---------------------------------------------------------------------------
// attention stage 2: flash-style softmax(QK^T + bias + mask) @ V per (i,h)
// ---------------------------------------------------------------------------
__global__ __launch_bounds__(256) void k_attn(
    const __bf16* __restrict__ qw, const __bf16* __restrict__ kw, const __bf16* __restrict__ vw,
    const float* __restrict__ tb, const float* __restrict__ mask,
    __bf16* __restrict__ ow, int transpose) {
  __shared__ __align__(32) __bf16 ks[256 * 32];
  __shared__ __align__(32) __bf16 vs[256 * 32];
  __shared__ __align__(32) __bf16 qs[128 * 32];
  __shared__ __align__(32) __bf16 pb[8][16 * 32];
  const int i = blockIdx.x, h = blockIdx.y;
  const int lane = threadIdx.x & 31, wave = threadIdx.x >> 5;
  const size_t headBase = (size_t)(i * 4 + h) * 8192;
  const int qOff = blockIdx.z * 128;
  // async stage K,V (16KB each) and the 8KB Q chunk into LDS
  for (int t = threadIdx.x; t < 1024; t += 256) {
    async_ld_b128(ks + t * 8, kw + headBase + t * 8);
    async_ld_b128(vs + t * 8, vw + headBase + t * 8);
  }
  for (int t = threadIdx.x; t < 512; t += 256)
    async_ld_b128(qs + t * 8, qw + headBase + (size_t)qOff * 32 + t * 8);
  wait_async0();
  __syncthreads();
  const int n = lane & 15, half = (lane >> 4) & 1;
  const int qBase = wave * 16;
  const int qRow0 = qOff + qBase;
  v8f o0 = {0.f, 0.f, 0.f, 0.f, 0.f, 0.f, 0.f, 0.f};
  v8f o1 = {0.f, 0.f, 0.f, 0.f, 0.f, 0.f, 0.f, 0.f};
  float mcur[8], lsum[8];
#pragma unroll
  for (int r = 0; r < 8; ++r) { mcur[r] = -1e30f; lsum[r] = 0.0f; }
  const float* tbh = tb + (size_t)h * 65536;
#pragma unroll 1
  for (int kt = 0; kt < 256; kt += 32) {
    const v8f z8 = {0.f, 0.f, 0.f, 0.f, 0.f, 0.f, 0.f, 0.f};
    v16bf aq = load_a(qs + qBase * 32, 32, 0, lane);
    v16bf b0 = load_b_bf16rows(ks, 32, kt, 0, lane);
    v16bf b1 = load_b_bf16rows(ks, 32, kt + 16, 0, lane);
    v8f s0 = wmma_bf16(aq, b0, z8);
    v8f s1 = wmma_bf16(aq, b1, z8);
    const int k0 = kt + n, k1 = kt + 16 + n;
    const float mb0 = 1e9f * ((transpose ? mask[(size_t)k0 * 256 + i] : mask[(size_t)i * 256 + k0]) - 1.0f);
    const float mb1 = 1e9f * ((transpose ? mask[(size_t)k1 * 256 + i] : mask[(size_t)i * 256 + k1]) - 1.0f);
#pragma unroll
    for (int r = 0; r < 8; ++r) {
      const int q = qRow0 + r + 8 * half;
      s0[r] += tbh[(size_t)q * 256 + k0] + mb0;
      s1[r] += tbh[(size_t)q * 256 + k1] + mb1;
    }
#pragma unroll
    for (int r = 0; r < 8; ++r) {
      float mx = fmaxf(s0[r], s1[r]);
#pragma unroll
      for (int off = 8; off >= 1; off >>= 1) mx = fmaxf(mx, __shfl_xor(mx, off));
      const float mnew = fmaxf(mcur[r], mx);
      const float corr = __expf(mcur[r] - mnew);
      mcur[r] = mnew;
      const float p0 = __expf(s0[r] - mnew);
      const float p1 = __expf(s1[r] - mnew);
      s0[r] = p0; s1[r] = p1;
      float ps = p0 + p1;
#pragma unroll
      for (int off = 8; off >= 1; off >>= 1) ps += __shfl_xor(ps, off);
      lsum[r] = lsum[r] * corr + ps;
      o0[r] *= corr; o1[r] *= corr;
    }
    // spill P (16x32) to per-wave LDS, reload in A-fragment layout
#pragma unroll
    for (int r = 0; r < 8; ++r) {
      const int m = r + 8 * half;
      pb[wave][m * 32 + n] = (__bf16)s0[r];
      pb[wave][m * 32 + 16 + n] = (__bf16)s1[r];
    }
    __builtin_amdgcn_wave_barrier();   // keep the compiler from hoisting the reload above the spill
    v16bf ap = load_a(pb[wave], 32, 0, lane);
    v16bf bv0 = load_b_v(vs, kt, 0, lane);
    v16bf bv1 = load_b_v(vs, kt, 16, lane);
    o0 = wmma_bf16(ap, bv0, o0);
    o1 = wmma_bf16(ap, bv1, o1);
  }
#pragma unroll
  for (int r = 0; r < 8; ++r) {
    const int q = qRow0 + r + 8 * half;
    const float inv = 1.0f / lsum[r];
    ow[headBase + (size_t)q * 32 + n] = (__bf16)(o0[r] * inv);
    ow[headBase + (size_t)q * 32 + 16 + n] = (__bf16)(o1[r] * inv);
  }
}

// ---------------------------------------------------------------------------
// attention stage 3: z += (o * gate) @ Wo^T  (optionally store transposed)
// ---------------------------------------------------------------------------
__global__ __launch_bounds__(256) void k_attn_out(
    const __bf16* __restrict__ ow, const __bf16* __restrict__ gs, const __bf16* __restrict__ wo,
    float* __restrict__ zc, int transpose) {
  __shared__ __align__(32) __bf16 xs[128][136];
  const int lane = threadIdx.x & 31, wave = threadIdx.x >> 5;
  const int rowBase = blockIdx.x * 128;
  for (int rr = wave; rr < 128; rr += 8) {
    const int r0 = rowBase + rr;
    const int i0 = r0 >> 8, j0 = r0 & 255;
#pragma unroll
    for (int t = 0; t < 4; ++t) {
      const int col = lane * 4 + t;
      const int h = col >> 5, dh = col & 31;
      const float val = (float)ow[((size_t)(i0 * 4 + h) * 256 + j0) * 32 + dh] *
                        (float)gs[(size_t)r0 * 128 + col];
      xs[rr][col] = (__bf16)val;
    }
  }
  __syncthreads();
  const int mBase = wave * 16;
  const int n = lane & 15, half = (lane >> 4) & 1;
#pragma unroll 1
  for (int c = 0; c < 8; ++c) {
    const int colBase = c * 16;
    v8f acc = gemm_k128(xs, mBase, wo, colBase, lane);
    const int d = colBase + n;
#pragma unroll
    for (int r = 0; r < 8; ++r) {
      const int r0 = rowBase + mBase + r + 8 * half;
      const int i0 = r0 >> 8, j0 = r0 & 255;
      const size_t idx = (size_t)(transpose ? (j0 * 256 + i0) : r0) * 128 + d;
      zc[idx] += acc[r];
    }
  }
}

// ---------------------------------------------------------------------------
// transition stage 1: h = silu(LN(z)@fc1^T) * (LN(z)@fc2^T)   [65536 x 512] bf16
// ---------------------------------------------------------------------------
__global__ __launch_bounds__(256) void k_trans1(
    const float* __restrict__ zc, const float* __restrict__ nw, const float* __restrict__ nb,
    const __bf16* __restrict__ fc1, const __bf16* __restrict__ fc2, __bf16* __restrict__ hw) {
  __shared__ __align__(32) __bf16 xs[128][136];
  const int lane = threadIdx.x & 31, wave = threadIdx.x >> 5;
  const int rowBase = blockIdx.x * 128;
  ln_rows_to_lds(zc, nw, nb, xs, rowBase, 0, lane, wave);
  __syncthreads();
  const int mBase = wave * 16;
  const int n = lane & 15, half = (lane >> 4) & 1;
#pragma unroll 1
  for (int c = 0; c < 32; ++c) {
    const int colBase = c * 16;
    v8f a1 = gemm_k128(xs, mBase, fc1, colBase, lane);
    v8f a2 = gemm_k128(xs, mBase, fc2, colBase, lane);
    const int col = colBase + n;
#pragma unroll
    for (int r = 0; r < 8; ++r) {
      const int r0 = rowBase + mBase + r + 8 * half;
      const float x1 = a1[r];
      hw[(size_t)r0 * 512 + col] = (__bf16)(x1 * sigmoidf_(x1) * a2[r]);
    }
  }
}

// ---------------------------------------------------------------------------
// transition stage 2: out = z + h @ fc3^T   (K = 512, chunked through LDS)
// ---------------------------------------------------------------------------
__global__ __launch_bounds__(256) void k_trans2(
    const __bf16* __restrict__ hw, const __bf16* __restrict__ fc3,
    const float* __restrict__ zc, float* __restrict__ out) {
  __shared__ __align__(32) __bf16 xs[128][136];
  const int lane = threadIdx.x & 31, wave = threadIdx.x >> 5;
  const int rowBase = blockIdx.x * 128;
  const int mBase = wave * 16;
  const int n = lane & 15, half = (lane >> 4) & 1;
  v8f acc[8];
#pragma unroll
  for (int c = 0; c < 8; ++c) acc[c] = (v8f){0.f, 0.f, 0.f, 0.f, 0.f, 0.f, 0.f, 0.f};
#pragma unroll 1
  for (int kb = 0; kb < 4; ++kb) {
    __syncthreads();
    // async stage 128x128 bf16 h-chunk: 1024 x 16B transfers
    for (int t = threadIdx.x; t < 1024; t += 256) {
      const int rr = t >> 3, cc = (t & 7) * 16;
      async_ld_b128(&xs[rr][cc], hw + (size_t)(rowBase + rr) * 512 + kb * 128 + cc);
    }
    wait_async0();
    __syncthreads();
#pragma unroll 1
    for (int c = 0; c < 8; ++c) {
      const int colBase = c * 16;
#pragma unroll
      for (int kk = 0; kk < 128; kk += 32) {
        v16bf a = load_a(&xs[mBase][0], 136, kk, lane);
        v16bf b = load_b_bf16rows(fc3, 512, colBase, kb * 128 + kk, lane);
        acc[c] = wmma_bf16(a, b, acc[c]);
      }
    }
  }
#pragma unroll 1
  for (int c = 0; c < 8; ++c) {
    const int d = c * 16 + n;
#pragma unroll
    for (int r = 0; r < 8; ++r) {
      const int r0 = rowBase + mBase + r + 8 * half;
      const size_t idx = (size_t)r0 * 128 + d;
      out[idx] = zc[idx] + acc[c][r];
    }
  }
}

// ---------------------------------------------------------------------------
// host-side launch
// ---------------------------------------------------------------------------
extern "C" void kernel_launch(void* const* d_in, const int* in_sizes, int n_in,
                              void* d_out, int out_size, void* d_ws, size_t ws_size,
                              hipStream_t stream) {
  (void)in_sizes; (void)n_in; (void)out_size; (void)ws_size;
  const float* z    = (const float*)d_in[0];
  const float* mask = (const float*)d_in[1];
  const float* tmo[8]; for (int k = 0; k < 8; ++k) tmo[k] = (const float*)d_in[2 + k];
  const float* tmi[8]; for (int k = 0; k < 8; ++k) tmi[k] = (const float*)d_in[10 + k];
  const float* tas[8]; for (int k = 0; k < 8; ++k) tas[k] = (const float*)d_in[18 + k];
  const float* tae[8]; for (int k = 0; k < 8; ++k) tae[k] = (const float*)d_in[26 + k];
  const float* trs[5]; for (int k = 0; k < 5; ++k) trs[k] = (const float*)d_in[34 + k];

  char* ws = (char*)d_ws;
  // workspace map (bytes):
  //   [0,   32M)   z_cur fp32
  //   [32M, 48M)   A: aT / q_ws   (transition: h_ws spans [32M, 96M))
  //   [48M, 64M)   B: bT / k_ws
  //   [64M, 80M)   C: gsig / v_ws
  //   [80M, 112M)  D: o fp32  (attn: gate bf16 [80M,96M) + o_ws bf16 [96M,112M))
  //   [112M,113M)  tb fp32
  //   [113M,~114M) bf16 weights
  float*  z_cur = (float*)(ws);
  __bf16* aT    = (__bf16*)(ws + 33554432ull);
  __bf16* bT    = (__bf16*)(ws + 50331648ull);
  __bf16* gsig  = (__bf16*)(ws + 67108864ull);
  float*  otmp  = (float*)(ws + 83886080ull);
  __bf16* qwk   = aT;
  __bf16* kwk   = bT;
  __bf16* vwk   = gsig;
  __bf16* gatt  = (__bf16*)(ws + 83886080ull);
  __bf16* owk   = (__bf16*)(ws + 100663296ull);
  float*  tbuf  = (float*)(ws + 117440512ull);
  __bf16* hbuf  = aT;  // [32M, 96M): free by transition time
  __bf16* wbf   = (__bf16*)(ws + 118489088ull);

  // bf16 weight pack: {tmo.p_in,g_in,p_out,g_out, tmi..., tas.bias,q,k,v,g,o, tae..., fc1,fc2,fc3}
  const float* wsrc[23] = { tmo[2], tmo[3], tmo[6], tmo[7],
                            tmi[2], tmi[3], tmi[6], tmi[7],
                            tas[2], tas[3], tas[4], tas[5], tas[6], tas[7],
                            tae[2], tae[3], tae[4], tae[5], tae[6], tae[7],
                            trs[2], trs[3], trs[4] };
  const unsigned wlen[23] = { 32768, 32768, 16384, 16384,
                              32768, 32768, 16384, 16384,
                              512, 16384, 16384, 16384, 16384, 16384,
                              512, 16384, 16384, 16384, 16384, 16384,
                              65536, 65536, 65536 };
  WPack pk;
  unsigned acc = 0;
  unsigned woff[23];
  for (int k = 0; k < 23; ++k) { pk.src[k] = wsrc[k]; pk.len[k] = wlen[k]; pk.off[k] = woff[k] = acc; acc += wlen[k]; }
  const __bf16* B_tmo_pin = wbf + woff[0],  *B_tmo_gin = wbf + woff[1];
  const __bf16* B_tmo_pout = wbf + woff[2], *B_tmo_gout = wbf + woff[3];
  const __bf16* B_tmi_pin = wbf + woff[4],  *B_tmi_gin = wbf + woff[5];
  const __bf16* B_tmi_pout = wbf + woff[6], *B_tmi_gout = wbf + woff[7];
  const __bf16* B_tas_bias = wbf + woff[8], *B_tas_q = wbf + woff[9], *B_tas_k = wbf + woff[10];
  const __bf16* B_tas_v = wbf + woff[11],   *B_tas_g = wbf + woff[12], *B_tas_o = wbf + woff[13];
  const __bf16* B_tae_bias = wbf + woff[14], *B_tae_q = wbf + woff[15], *B_tae_k = wbf + woff[16];
  const __bf16* B_tae_v = wbf + woff[17],   *B_tae_g = wbf + woff[18], *B_tae_o = wbf + woff[19];
  const __bf16* B_fc1 = wbf + woff[20], *B_fc2 = wbf + woff[21], *B_fc3 = wbf + woff[22];

  dim3 blk(256);
  k_cvt_w<<<dim3(64, 23), blk, 0, stream>>>(pk, wbf);
  // --- triangle multiplication (outgoing) ---
  k_trimult_in <<<512, blk, 0, stream>>>(z, mask, tmo[0], tmo[1], B_tmo_pin, B_tmo_gin, B_tmo_gout, aT, bT, gsig, 0);
  k_tri_gemm   <<<dim3(128, 16), blk, 0, stream>>>(aT, bT, otmp);
  k_trimult_out<<<512, blk, 0, stream>>>(otmp, gsig, tmo[4], tmo[5], B_tmo_pout, z, z_cur);
  // --- triangle multiplication (incoming) ---
  k_trimult_in <<<512, blk, 0, stream>>>(z_cur, mask, tmi[0], tmi[1], B_tmi_pin, B_tmi_gin, B_tmi_gout, aT, bT, gsig, 1);
  k_tri_gemm   <<<dim3(128, 16), blk, 0, stream>>>(aT, bT, otmp);
  k_trimult_out<<<512, blk, 0, stream>>>(otmp, gsig, tmi[4], tmi[5], B_tmi_pout, z_cur, z_cur);
  // --- triangle attention (starting node) ---
  k_attn_qkvg<<<512, blk, 0, stream>>>(z_cur, tas[0], tas[1], B_tas_q, B_tas_k, B_tas_v, B_tas_g, B_tas_bias,
                                       qwk, kwk, vwk, gatt, tbuf, 0);
  k_attn     <<<dim3(256, 4, 2), blk, 0, stream>>>(qwk, kwk, vwk, tbuf, mask, owk, 0);
  k_attn_out <<<512, blk, 0, stream>>>(owk, gatt, B_tas_o, z_cur, 0);
  // --- triangle attention (ending node) ---
  k_attn_qkvg<<<512, blk, 0, stream>>>(z_cur, tae[0], tae[1], B_tae_q, B_tae_k, B_tae_v, B_tae_g, B_tae_bias,
                                       qwk, kwk, vwk, gatt, tbuf, 1);
  k_attn     <<<dim3(256, 4, 2), blk, 0, stream>>>(qwk, kwk, vwk, tbuf, mask, owk, 1);
  k_attn_out <<<512, blk, 0, stream>>>(owk, gatt, B_tae_o, z_cur, 1);
  // --- transition ---
  k_trans1<<<512, blk, 0, stream>>>(z_cur, trs[0], trs[1], B_fc1, B_fc2, hbuf);
  k_trans2<<<512, blk, 0, stream>>>(hbuf, B_fc3, z_cur, (float*)d_out);
}
